// PlasticityModelMoE_31155692765804
// MI455X (gfx1250) — compile-verified, matrix operand order checked by reading, not run on hardware
//
#include <hip/hip_runtime.h>
#include <hip/hip_bf16.h>

typedef __attribute__((ext_vector_type(16))) _Float16 v16h;
typedef __attribute__((ext_vector_type(8)))  float    v8f;

// Problem sizes (fixed by the reference)
#define NN  2048
#define DD  1024
#define UU  2048
#define NB  4
#define MM  8192
#define MD  1024

// ---------------- workspace layout (bytes) ----------------
#define OFF_X16    ((size_t)0)                         // [N,D]  f16   4.19 MB
#define OFF_WMODT  ((size_t)4194304)                   // [U*NB, D] f16 16.78 MB
#define OFF_RWT    ((size_t)20971520)                  // [M, U] f16   33.55 MB
#define OFF_MEMT   ((size_t)54525952)                  // [MD, M] f16  16.78 MB
#define OFF_GATE   ((size_t)71303168)                  // [N, NB] f32
#define OFF_CONN   ((size_t)71335936)                  // [U] f32
#define OFF_BLEND  ((size_t)71344128)                  // [N, U] f16   8.39 MB
#define OFF_BIG    ((size_t)79732736)                  // branch f16 then logits f32 (67.1 MB)
#define OFF_ATTN   ((size_t)146841600)                 // [N, M] f16   33.55 MB

__device__ __forceinline__ float sigmoidf_(float x) { return 1.0f / (1.0f + __expf(-x)); }

// ---------------- conversion kernels ----------------
__global__ void k_cvt_x(const float* __restrict__ x, _Float16* __restrict__ out) {
  size_t i = (size_t)blockIdx.x * 256 + threadIdx.x;
  out[i] = (_Float16)x[i];
}

// LDS-tiled transpose+convert: out[c][r] = f16(in[r][c]); in is [R,C] row-major.
// 64x64 tile, 256 threads; LDS stride 66 halves (33 dwords, odd) -> conflict-free.
__global__ void k_tcvt_t(const float* __restrict__ in, _Float16* __restrict__ out, int R, int C) {
  __shared__ _Float16 tile[64 * 66];
  const int t = threadIdx.x;
  const int ci = t & 63, r0 = t >> 6;                 // r0 in 0..3
  const int cBase = blockIdx.x * 64, rBase = blockIdx.y * 64;
#pragma unroll 4
  for (int i = 0; i < 16; ++i) {
    int r = r0 + i * 4;
    tile[r * 66 + ci] = (_Float16)in[(size_t)(rBase + r) * C + cBase + ci];
  }
  __syncthreads();
#pragma unroll 4
  for (int i = 0; i < 16; ++i) {
    int oc = r0 + i * 4;                              // output row = original column
    out[(size_t)(cBase + oc) * R + rBase + ci] = tile[ci * 66 + oc];
  }
}

// Fused: wmodT[(u*4+b)][d] = f16( w[d][u*4+b] * sigmoid(delay[d][u*4+b]) ), tiled transpose
__global__ void k_wmod_t(const float* __restrict__ w, const float* __restrict__ delay,
                         _Float16* __restrict__ outT) {
  __shared__ _Float16 tile[64 * 66];
  const int t = threadIdx.x;
  const int ci = t & 63, r0 = t >> 6;
  const int cBase = blockIdx.x * 64, rBase = blockIdx.y * 64;   // C = U*NB, R = D
#pragma unroll 4
  for (int i = 0; i < 16; ++i) {
    int r = r0 + i * 4;
    size_t idx = (size_t)(rBase + r) * (UU * NB) + cBase + ci;
    tile[r * 66 + ci] = (_Float16)(w[idx] * sigmoidf_(delay[idx]));
  }
  __syncthreads();
#pragma unroll 4
  for (int i = 0; i < 16; ++i) {
    int oc = r0 + i * 4;
    outT[(size_t)(cBase + oc) * DD + rBase + ci] = tile[ci * 66 + oc];
  }
}

// ---------------- gate: softmax(x @ gate_W + gate_b) over NB=4, wave-per-row ----------------
__global__ void k_gate(const float* __restrict__ x, const float* __restrict__ gW,
                       const float* __restrict__ gb, float* __restrict__ gate) {
  int lane = threadIdx.x & 31, wave = threadIdx.x >> 5;
  int n = blockIdx.x * 8 + wave;
  float s0 = 0.f, s1 = 0.f, s2 = 0.f, s3 = 0.f;
  const float* xr = x + (size_t)n * DD;
  for (int d = lane; d < DD; d += 32) {
    float xv = xr[d];
    const float* g = gW + (size_t)d * NB;
    s0 += xv * g[0]; s1 += xv * g[1]; s2 += xv * g[2]; s3 += xv * g[3];
  }
  for (int m = 16; m >= 1; m >>= 1) {
    s0 += __shfl_xor(s0, m, 32); s1 += __shfl_xor(s1, m, 32);
    s2 += __shfl_xor(s2, m, 32); s3 += __shfl_xor(s3, m, 32);
  }
  s0 += gb[0]; s1 += gb[1]; s2 += gb[2]; s3 += gb[3];
  float mx = fmaxf(fmaxf(s0, s1), fmaxf(s2, s3));
  float e0 = __expf(s0 - mx), e1 = __expf(s1 - mx), e2 = __expf(s2 - mx), e3 = __expf(s3 - mx);
  float inv = 1.0f / (e0 + e1 + e2 + e3);
  if (lane == 0) {
    float* gr = gate + (size_t)n * NB;
    gr[0] = e0 * inv; gr[1] = e1 * inv; gr[2] = e2 * inv; gr[3] = e3 * inv;
  }
}

// ---------------- conn = sigmoid(relu(navg @ W1 + b1) @ W2 + b2), one block ----------------
__global__ void k_conn(const float* __restrict__ navg, const float* __restrict__ W1,
                       const float* __restrict__ b1, const float* __restrict__ W2,
                       const float* __restrict__ b2, float* __restrict__ conn) {
  __shared__ float h[32];
  int t = threadIdx.x;
  if (t < 32) {
    float s = b1[t];
    for (int u = 0; u < UU; ++u) s += navg[u] * W1[(size_t)u * 32 + t];
    h[t] = fmaxf(s, 0.f);
  }
  __syncthreads();
  for (int u = t; u < UU; u += 256) {
    float s = b2[u];
    for (int j = 0; j < 32; ++j) s += h[j] * W2[(size_t)j * UU + u];
    conn[u] = sigmoidf_(s);
  }
}

// ---------------- WMMA GEMM: C[M,N] = A[M,K] * BT[N,K]^T, f16 in, f32/f16 out ----------------
// Double-buffered LDS, 128x128 block tile, BK=32, 8 waves each owning a 32x64 sub-tile.
__device__ __forceinline__ v16h lds_frag(const _Float16* tile, int row, int lane) {
  // 16-bit A/B fragment: lane holds row/col (lane&15); K chunks: lanes0-15 -> {0,16}, lanes16-31 -> {8,24}
  const _Float16* p = tile + row * 32 + ((lane & 16) ? 8 : 0);
  union { v16h v; uint4 q[2]; } u;
  u.q[0] = *(const uint4*)(p);
  u.q[1] = *(const uint4*)(p + 16);
  return u.v;
}

template <class CT>
__global__ void k_gemm(const _Float16* __restrict__ A, const _Float16* __restrict__ BT,
                       CT* __restrict__ C, int Mdim, int Ndim, int Kdim) {
  __shared__ alignas(16) _Float16 As[2][128 * 32];
  __shared__ alignas(16) _Float16 Bs[2][128 * 32];
  const int t = threadIdx.x, lane = t & 31, wave = t >> 5;
  const int wm = wave & 3;       // 32-row slab within block tile
  const int wn = wave >> 2;      // 64-col slab within block tile
  const int rowBase = blockIdx.y * 128;
  const int colBase = blockIdx.x * 128;
  const int lr = t >> 1, lo = (t & 1) * 16;   // staging: each thread moves 32B of a 64B row

  const _Float16* gA = A + (size_t)(rowBase + lr) * Kdim + lo;
  const _Float16* gB = BT + (size_t)(colBase + lr) * Kdim + lo;

  auto stage = [&](int buf, int k0) {
    const uint4* ga = (const uint4*)(gA + k0);
    uint4* sa = (uint4*)(As[buf] + lr * 32 + lo);
    sa[0] = ga[0]; sa[1] = ga[1];
    const uint4* gb = (const uint4*)(gB + k0);
    uint4* sb = (uint4*)(Bs[buf] + lr * 32 + lo);
    sb[0] = gb[0]; sb[1] = gb[1];
  };

  v8f acc[2][4] = {};
  const int nIter = Kdim >> 5;

  stage(0, 0);
  __syncthreads();

  for (int it = 0; it < nIter; ++it) {
    const int buf = it & 1;
    if (it + 1 < nIter) {
      stage(buf ^ 1, (it + 1) << 5);              // overlap next-tile staging with WMMA
      if (it + 2 < nIter) {                        // near-cache prefetch of tile after next
        __builtin_prefetch(gA + ((it + 2) << 5), 0, 3);
        __builtin_prefetch(gB + ((it + 2) << 5), 0, 3);
      }
    }
    v16h af[2], bf[4];
#pragma unroll
    for (int i = 0; i < 2; ++i) af[i] = lds_frag(As[buf], wm * 32 + i * 16 + (lane & 15), lane);
#pragma unroll
    for (int j = 0; j < 4; ++j) bf[j] = lds_frag(Bs[buf], wn * 64 + j * 16 + (lane & 15), lane);
#pragma unroll
    for (int i = 0; i < 2; ++i)
#pragma unroll
      for (int j = 0; j < 4; ++j)
        acc[i][j] = __builtin_amdgcn_wmma_f32_16x16x32_f16(
            false, af[i], false, bf[j], (short)0, acc[i][j], false, false);
    __syncthreads();
  }

#pragma unroll
  for (int i = 0; i < 2; ++i)
#pragma unroll
    for (int j = 0; j < 4; ++j) {
      int mb = rowBase + wm * 32 + i * 16 + ((lane >> 4) << 3);
      int nb = colBase + wn * 64 + j * 16 + (lane & 15);
#pragma unroll
      for (int r = 0; r < 8; ++r)
        C[(size_t)(mb + r) * Ndim + nb] = (CT)acc[i][j][r];
    }
}

// ---------------- combine: z = sum_b (branch+bias)*gate; a=relu(z*conn*mask); blend ----------------
__global__ void k_combine(const _Float16* __restrict__ branch, const float* __restrict__ gate,
                          const float* __restrict__ bias, const float* __restrict__ conn,
                          const float* __restrict__ mask, const float* __restrict__ act_w,
                          _Float16* __restrict__ blend16) {
  size_t i = (size_t)blockIdx.x * 256 + threadIdx.x;   // over N*U
  int n = (int)(i >> 11);     // / 2048
  int u = (int)(i & 2047);
  const float* g = gate + (size_t)n * NB;
  const _Float16* br = branch + (size_t)n * (UU * NB) + (size_t)u * NB;
  const float* bb = bias + (size_t)u * NB;
  float z = 0.f;
#pragma unroll
  for (int b = 0; b < NB; ++b) z += ((float)br[b] + bb[b]) * g[b];
  float a = fmaxf(z * conn[u] * mask[u], 0.f);

  // softmax over 9 activation weights (entry 8 = zero activation)
  float aw[9], mx = act_w[0];
#pragma unroll
  for (int k = 0; k < 9; ++k) { aw[k] = act_w[k]; mx = fmaxf(mx, aw[k]); }
  float s = 0.f;
#pragma unroll
  for (int k = 0; k < 9; ++k) { aw[k] = __expf(aw[k] - mx); s += aw[k]; }
  float inv = 1.0f / s;

  float sg   = sigmoidf_(a);
  float elu  = (a > 0.f) ? a : expm1f(a);
  float th   = tanhf(a);
  float rl   = a;                        // a >= 0 already
  float si   = a * sg;
  float ge   = 0.5f * a * (1.0f + erff(a * 0.70710678118654752f));
  float se   = 1.0507009873554805f * ((a > 0.f) ? a : 1.6732632423543772f * expm1f(a));
  float sp   = a + log1pf(__expf(-a));   // softplus, stable for a >= 0
  float mi   = a * tanhf(sp);

  float out = (aw[0] * sg + aw[1] * elu + aw[2] * th + aw[3] * rl +
               aw[4] * si + aw[5] * ge + aw[6] * se + aw[7] * mi) * inv;
  blend16[i] = (_Float16)out;
}

// ---------------- row softmax over M=8192: attn16 = softmax(logits + read_b) ----------------
__global__ void k_softmax(const float* __restrict__ logits, const float* __restrict__ rb,
                          _Float16* __restrict__ attn16) {
  __shared__ float red[256];
  int t = threadIdx.x, n = blockIdx.x;
  const float* L = logits + (size_t)n * MM;
  float m = -3.4e38f;
  for (int i = t; i < MM; i += 256) m = fmaxf(m, L[i] + rb[i]);
  red[t] = m; __syncthreads();
  for (int s = 128; s >= 1; s >>= 1) { if (t < s) red[t] = fmaxf(red[t], red[t + s]); __syncthreads(); }
  float M = red[0]; __syncthreads();
  float acc = 0.f;
  for (int i = t; i < MM; i += 256) acc += __expf(L[i] + rb[i] - M);
  red[t] = acc; __syncthreads();
  for (int s = 128; s >= 1; s >>= 1) { if (t < s) red[t] += red[t + s]; __syncthreads(); }
  float inv = 1.0f / red[0];
  _Float16* A = attn16 + (size_t)n * MM;
  for (int i = t; i < MM; i += 256) A[i] = (_Float16)(__expf(L[i] + rb[i] - M) * inv);
}

// ---------------- launch ----------------
extern "C" void kernel_launch(void* const* d_in, const int* in_sizes, int n_in,
                              void* d_out, int out_size, void* d_ws, size_t ws_size,
                              hipStream_t stream) {
  const float* x      = (const float*)d_in[0];
  const float* w      = (const float*)d_in[1];
  const float* delay  = (const float*)d_in[2];
  const float* bias   = (const float*)d_in[3];
  const float* gate_W = (const float*)d_in[4];
  const float* gate_b = (const float*)d_in[5];
  const float* navg   = (const float*)d_in[6];
  const float* cW1    = (const float*)d_in[7];
  const float* cb1    = (const float*)d_in[8];
  const float* cW2    = (const float*)d_in[9];
  const float* cb2    = (const float*)d_in[10];
  const float* mask   = (const float*)d_in[11];
  const float* act_w  = (const float*)d_in[12];
  const float* read_W = (const float*)d_in[13];
  const float* read_b = (const float*)d_in[14];
  const float* memory = (const float*)d_in[15];
  float* out = (float*)d_out;

  char* ws = (char*)d_ws;
  _Float16* x16    = (_Float16*)(ws + OFF_X16);
  _Float16* wmodT  = (_Float16*)(ws + OFF_WMODT);
  _Float16* rwT    = (_Float16*)(ws + OFF_RWT);
  _Float16* memT   = (_Float16*)(ws + OFF_MEMT);
  float*    gate   = (float*)   (ws + OFF_GATE);
  float*    conn   = (float*)   (ws + OFF_CONN);
  _Float16* blend  = (_Float16*)(ws + OFF_BLEND);
  _Float16* branch = (_Float16*)(ws + OFF_BIG);    // f16 branch ...
  float*    logits = (float*)   (ws + OFF_BIG);    // ... then overwritten by f32 logits
  _Float16* attn   = (_Float16*)(ws + OFF_ATTN);

  // conversions / transposes (coalesced, LDS-tiled)
  k_cvt_x<<<(NN * DD) / 256, 256, 0, stream>>>(x, x16);
  {
    dim3 g((UU * NB) / 64, DD / 64);
    k_wmod_t<<<g, 256, 0, stream>>>(w, delay, wmodT);
  }
  {
    dim3 g(MM / 64, UU / 64);                        // read_W [U, M] -> rwT [M, U]
    k_tcvt_t<<<g, 256, 0, stream>>>(read_W, rwT, UU, MM);
  }
  {
    dim3 g(MD / 64, MM / 64);                        // memory [M, MD] -> memT [MD, M]
    k_tcvt_t<<<g, 256, 0, stream>>>(memory, memT, MM, MD);
  }

  // small nets
  k_gate<<<NN / 8, 256, 0, stream>>>(x, gate_W, gate_b, gate);
  k_conn<<<1, 256, 0, stream>>>(navg, cW1, cb1, cW2, cb2, conn);

  // GEMM 1: branch[N, U*NB] = x16 @ wmodT^T   (K = D)
  {
    dim3 grid((UU * NB) / 128, NN / 128);
    k_gemm<_Float16><<<grid, 256, 0, stream>>>(x16, wmodT, branch, NN, UU * NB, DD);
  }

  // combine -> blend16[N,U]
  k_combine<<<((size_t)NN * UU) / 256, 256, 0, stream>>>(branch, gate, bias, conn, mask, act_w, blend);

  // GEMM 2: logits[N, M] = blend @ rwT^T      (K = U)
  {
    dim3 grid(MM / 128, NN / 128);
    k_gemm<float><<<grid, 256, 0, stream>>>(blend, rwT, logits, NN, MM, UU);
  }

  // softmax rows -> attn16[N,M]
  k_softmax<<<NN, 256, 0, stream>>>(logits, read_b, attn);

  // GEMM 3: out[N, MD] = attn @ memT^T        (K = M)
  {
    dim3 grid(MD / 128, NN / 128);
    k_gemm<float><<<grid, 256, 0, stream>>>(attn, memT, out, NN, MD, MM);
  }
}
